// Block_2207613190325
// MI455X (gfx1250) — matrix operand order repaired and checked by reading, not compiled, not run on hardware
//
#include <hip/hip_runtime.h>

// ---------------------------------------------------------------------------
// CDNA5 / gfx1250 transformer block.
//  * All GEMMs + attention on v_wmma_f32_16x16x32_bf16 (fp32 accumulate):
//    compute-bound workload (AI ~2.4 kFLOP/B), bf16 WMMA has 8x the K-depth
//    of the f32 WMMA path.
//  * Tile staging via the Tensor Data Mover (tensor_load_to_lds + TENSORcnt),
//    with D# LDS padding generating bank-conflict-free pitches.
//  * Weights stored N-major (pre-transposed) so both GEMM operands are plain
//    2-D TDM tiles.
// ---------------------------------------------------------------------------

typedef __bf16 bf16;
typedef bf16  v16bf __attribute__((ext_vector_type(16)));
typedef bf16  v8bf  __attribute__((ext_vector_type(8)));
typedef float v8f   __attribute__((ext_vector_type(8)));
typedef unsigned int u32x4 __attribute__((ext_vector_type(4)));
typedef int          i32x4 __attribute__((ext_vector_type(4)));
typedef int          i32x8 __attribute__((ext_vector_type(8)));

#define GEMM_PITCH 40   // bf16 elems: 32 data + 8 pad (80B rows, 16B aligned)
#define ATTN_PITCH 72   // bf16 elems: 64 data + 8 pad (144B rows)

__device__ inline v8f zero8() {
    v8f z;
#pragma unroll
    for (int i = 0; i < 8; ++i) z[i] = 0.0f;
    return z;
}

__device__ inline v8f wmma_bf16(v16bf a, v16bf b, v8f c) {
    return __builtin_amdgcn_wmma_f32_16x16x32_bf16(
        false, a, false, b, (short)0, c, false, false);
}

// ---------------------------------------------------------------------------
// Tensor Data Mover: 2-D bf16 tile (tile_d0 elems/row x tile_d1 rows) from
// global (row stride `stride_elems`) into LDS at byte offset `lds_off`,
// padding each row of tile_d0 elems with 8 elems (4 DWORDs) to reach the
// bank-staggered LDS pitch.  pad_interval code: 3 -> 16 DW (64B row),
// 4 -> 32 DW (128B row).  D# bitfields per CDNA5 ISA 8.3/8.4.
// ---------------------------------------------------------------------------
__device__ inline void tdm_load_2d(unsigned lds_off, const void* gptr,
                                   unsigned tile_d0, unsigned tile_d1,
                                   unsigned long long stride_elems,
                                   int pad_interval_code) {
    unsigned long long ga = (unsigned long long)(size_t)gptr;
    u32x4 g0;
    g0[0] = 1u;                                    // count=1 (valid), user mode
    g0[1] = lds_off;                               // lds_addr (bytes)
    g0[2] = (unsigned)(ga & 0xffffffffu);          // global_addr[31:0]
    g0[3] = (unsigned)((ga >> 32) & 0x01ffffffu)   // global_addr[56:32]
          | (2u << 30);                            // type = 2 ("image")
    i32x8 g1;
    // word0: wg_mask=0 | data_size=1 (2B) | pad_enable | pad_interval | pad_amount=3 (4 DW)
    g1[0] = (1 << 16) | (1 << 20) | (pad_interval_code << 22) | (3 << 25);
    // tensor_dim0 = tile_d0 (tile fully in-bounds), bits[79:48]
    g1[1] = (int)((tile_d0 & 0xffffu) << 16);
    // word2: tensor_dim0[31:16] | tensor_dim1[15:0]<<16   (tensor_dim1 = tile_d1)
    g1[2] = (int)(((tile_d0 >> 16) & 0xffffu) | ((tile_d1 & 0xffffu) << 16));
    // word3: tensor_dim1[31:16] | tile_dim0<<16
    g1[3] = (int)(((tile_d1 >> 16) & 0xffffu) | ((tile_d0 & 0xffffu) << 16));
    // word4: tile_dim1 | tile_dim2=0
    g1[4] = (int)(tile_d1 & 0xffffu);
    // word5/6: tensor_dim0_stride (48b), tensor_dim1_stride=0
    g1[5] = (int)(stride_elems & 0xffffffffu);
    g1[6] = (int)((stride_elems >> 32) & 0xffffu);
    g1[7] = 0;
    i32x4 z4 = {0, 0, 0, 0};
#if defined(__clang_major__) && (__clang_major__ >= 23)
    i32x8 z8 = {0, 0, 0, 0, 0, 0, 0, 0};
    __builtin_amdgcn_tensor_load_to_lds(g0, g1, z4, z4, z8, 0);
#else
    __builtin_amdgcn_tensor_load_to_lds(g0, g1, z4, z4, 0);
#endif
}

// A-fragment (16x32 bf16) from row-major LDS tile; ISA 16-bit A layout.
__device__ inline v16bf load_afrag(const bf16* lds, int rowbase, int pitch, int kbase) {
    const int lane = threadIdx.x & 31;
    const int half = lane >> 4;
    const bf16* p  = lds + (size_t)(rowbase + (lane & 15)) * pitch + kbase;
    v8bf a0 = *(const v8bf*)(p + 8 * half);
    v8bf a1 = *(const v8bf*)(p + 16 + 8 * half);
    v16bf a;
#pragma unroll
    for (int i = 0; i < 8; ++i) { a[i] = a0[i]; a[i + 8] = a1[i]; }
    return a;
}

// B-fragment (32x16 bf16) from N-major LDS tile (lds[n][k]); ISA B layout.
__device__ inline v16bf load_bfrag(const bf16* lds, int nbase, int pitch, int kbase) {
    const int lane = threadIdx.x & 31;
    const int half = lane >> 4;
    const bf16* p  = lds + (size_t)(nbase + (lane & 15)) * pitch + kbase + 8 * half;
    v8bf b0 = *(const v8bf*)(p);
    v8bf b1 = *(const v8bf*)(p + 16);
    v16bf b;
#pragma unroll
    for (int i = 0; i < 8; ++i) { b[i] = b0[i]; b[i + 8] = b1[i]; }
    return b;
}

// ---------------------------------------------------------------------------
// Weight conversion (one-time per launch, deterministic)
// ---------------------------------------------------------------------------
// wqkvT[n][c], n = sel*1024 + h*64 + d  (N-major for TDM B-tiles)
__global__ __launch_bounds__(256) void cvt_qkv_kernel(
    const float* __restrict__ wq, const float* __restrict__ wk, const float* __restrict__ wv,
    const float* __restrict__ bq, const float* __restrict__ bk, const float* __restrict__ bv,
    bf16* __restrict__ wqkvT, float* __restrict__ bqkv) {
    size_t idx = (size_t)blockIdx.x * 256 + threadIdx.x;   // n*1024 + c
    int n = (int)(idx >> 10);
    int c = (int)(idx & 1023);
    int sel = n >> 10, nn = n & 1023;
    int h = nn >> 6, d = nn & 63;
    const float* w = (sel == 0) ? wq : (sel == 1) ? wk : wv;
    wqkvT[idx] = (bf16)w[((size_t)h * 1024 + c) * 64 + d];
    if (c == 0) {
        const float* bb = (sel == 0) ? bq : (sel == 1) ? bk : bv;
        bqkv[n] = bb[h * 64 + d];
    }
}

// out[n][k] = in[k][n]  (fp32 [K,N] row-major -> bf16 [N,K] row-major)
__global__ __launch_bounds__(256) void cvt_bf16T_kernel(
    const float* __restrict__ in, bf16* __restrict__ out, int N, int K) {
    size_t idx = (size_t)blockIdx.x * 256 + threadIdx.x;   // n*K + k
    int n = (int)(idx / K);
    int k = (int)(idx % K);
    out[idx] = (bf16)in[(size_t)k * N + n];
}

// ---------------------------------------------------------------------------
// LayerNorm fp32 -> bf16, one 1024-row per 256-thread block
// ---------------------------------------------------------------------------
__global__ __launch_bounds__(256) void ln_bf16_kernel(
    const float* __restrict__ x, const float* __restrict__ g,
    const float* __restrict__ be, bf16* __restrict__ out) {
    const int C = 1024;
    const int row = blockIdx.x, tid = threadIdx.x;
    const float* xr = x + (size_t)row * C;
    float v[4], s = 0.0f;
#pragma unroll
    for (int j = 0; j < 4; ++j) { v[j] = xr[tid + j * 256]; s += v[j]; }
    __shared__ float red[256];
    red[tid] = s; __syncthreads();
#pragma unroll
    for (int off = 128; off > 0; off >>= 1) {
        if (tid < off) red[tid] += red[tid + off];
        __syncthreads();
    }
    float mean = red[0] * (1.0f / 1024.0f);
    __syncthreads();
    float vs = 0.0f;
#pragma unroll
    for (int j = 0; j < 4; ++j) { float d = v[j] - mean; vs += d * d; }
    red[tid] = vs; __syncthreads();
#pragma unroll
    for (int off = 128; off > 0; off >>= 1) {
        if (tid < off) red[tid] += red[tid + off];
        __syncthreads();
    }
    float rstd = rsqrtf(red[0] * (1.0f / 1024.0f) + 1e-5f);
#pragma unroll
    for (int j = 0; j < 4; ++j) {
        int c = tid + j * 256;
        out[(size_t)row * C + c] = (bf16)((v[j] - mean) * rstd * g[c] + be[c]);
    }
}

// ---------------------------------------------------------------------------
// GEMM: out[M,N] = A[M,K] * BT[N,K]^T (+bias)(+res)(relu)
// 128x256 block tile, 8 waves (2M x 4N), 64x64 per wave (4x4 WMMA acc).
// A/B tiles DMA'd by the TDM; main loop is pure ds_load + wmma.
// ---------------------------------------------------------------------------
template <bool OUT_BF16, bool RELU, bool RES>
__global__ __launch_bounds__(256) void gemm_bf16_kernel(
    const bf16* __restrict__ A, const bf16* __restrict__ BT,
    const float* __restrict__ bias, const float* __restrict__ res,
    void* __restrict__ outp, int M, int N, int K) {
    __shared__ __align__(16) bf16 ldsA[128 * GEMM_PITCH];
    __shared__ __align__(16) bf16 ldsB[256 * GEMM_PITCH];

    const int tid  = threadIdx.x;
    const int wave = tid >> 5;
    const int lane = tid & 31;
    const int half = lane >> 4;
    const int ln   = lane & 15;
    const int mblk = blockIdx.x * 128;
    const int nblk = blockIdx.y * 256;
    const int wm   = (wave & 1) * 64;
    const int wn   = (wave >> 1) * 64;

    const unsigned ldsA_off = (unsigned)(size_t)(void*)ldsA;
    const unsigned ldsB_off = (unsigned)(size_t)(void*)ldsB;

    v8f acc[4][4];
#pragma unroll
    for (int mt = 0; mt < 4; ++mt)
#pragma unroll
        for (int nt = 0; nt < 4; ++nt) acc[mt][nt] = zero8();

    for (int k0 = 0; k0 < K; k0 += 32) {
        __syncthreads();                    // previous slab consumed
        if (wave == 0) {
            tdm_load_2d(ldsA_off, A  + (size_t)mblk * K + k0, 32, 128,
                        (unsigned long long)K, /*16 DW rows*/ 3);
            tdm_load_2d(ldsB_off, BT + (size_t)nblk * K + k0, 32, 256,
                        (unsigned long long)K, 3);
            __builtin_amdgcn_s_wait_tensorcnt((short)0);
        }
        __syncthreads();                    // slab visible to all waves

        v16bf af[4];
#pragma unroll
        for (int mt = 0; mt < 4; ++mt)
            af[mt] = load_afrag(ldsA, wm + mt * 16, GEMM_PITCH, 0);
#pragma unroll
        for (int nt = 0; nt < 4; ++nt) {
            v16bf bfrag = load_bfrag(ldsB, wn + nt * 16, GEMM_PITCH, 0);
#pragma unroll
            for (int mt = 0; mt < 4; ++mt)
                acc[mt][nt] = wmma_bf16(af[mt], bfrag, acc[mt][nt]);
        }
    }

#pragma unroll
    for (int mt = 0; mt < 4; ++mt) {
#pragma unroll
        for (int nt = 0; nt < 4; ++nt) {
            const int gn = nblk + wn + nt * 16 + ln;
            const float bv = bias ? bias[gn] : 0.0f;
#pragma unroll
            for (int i = 0; i < 8; ++i) {
                const int gm = mblk + wm + mt * 16 + i + 8 * half;
                float v = acc[mt][nt][i] + bv;
                if (RES)  v += res[(size_t)gm * N + gn];
                if (RELU) v = v > 0.0f ? v : 0.0f;
                if (OUT_BF16) ((bf16*)outp)[(size_t)gm * N + gn] = (bf16)v;
                else          ((float*)outp)[(size_t)gm * N + gn] = v;
            }
        }
    }
}

// ---------------------------------------------------------------------------
// Causal flash attention; qkv bf16 [B*T, 3072], out bf16 [B*T, 1024].
// One (b,h) x 128-query tile per block; K tile staged by TDM each iteration.
// ---------------------------------------------------------------------------
__global__ __launch_bounds__(256) void attn_kernel(
    const bf16* __restrict__ qkv, bf16* __restrict__ out) {
    const int T = 2048, LD = 3072;
    const int bh = blockIdx.y;
    const int b  = bh >> 4;
    const int h  = bh & 15;
    const int qbase = blockIdx.x * 128;

    __shared__ __align__(16) bf16 ldsQ[128 * ATTN_PITCH];
    __shared__ __align__(16) bf16 ldsK[64 * ATTN_PITCH];
    __shared__ __align__(16) bf16 ldsV[64 * ATTN_PITCH];   // transposed [d][key]
    __shared__ __align__(16) bf16 ldsP[8 * 16 * ATTN_PITCH];

    const int tid  = threadIdx.x;
    const int wave = tid >> 5;
    const int lane = tid & 31;
    const int half = lane >> 4;
    const int ln   = lane & 15;

    const bf16* qptr = qkv + (size_t)(b * T + qbase) * LD + h * 64;
    const bf16* kptr = qkv + (size_t)(b * T) * LD + 1024 + h * 64;
    const bf16* vptr = qkv + (size_t)(b * T) * LD + 2048 + h * 64;
    const unsigned ldsK_off = (unsigned)(size_t)(void*)ldsK;

    // stage Q tile [128 x 64] once
    {
        const int r = tid >> 1, c0 = (tid & 1) * 32;
        const bf16* g = qptr + (size_t)r * LD + c0;
#pragma unroll
        for (int j = 0; j < 4; ++j)
            *(v8bf*)(ldsQ + (size_t)r * ATTN_PITCH + c0 + j * 8) =
                *(const v8bf*)(g + j * 8);
    }

    float m_i[8], l_i[8];
    v8f accO[4];
#pragma unroll
    for (int i = 0; i < 8; ++i) { m_i[i] = -1e30f; l_i[i] = 0.0f; }
#pragma unroll
    for (int dt = 0; dt < 4; ++dt) accO[dt] = zero8();

    const int nkt = (qbase >> 6) + 2;
    const float scale = 0.03125f;          // C^-0.5 = 1/32 per reference

    for (int kt = 0; kt < nkt; ++kt) {
        __syncthreads();
        // V tile transposed into LDS (TDM cannot transpose)
        {
            const int d = tid >> 2, key0 = (tid & 3) * 16;
            bf16 tmp[16];
#pragma unroll
            for (int i = 0; i < 16; ++i)
                tmp[i] = vptr[(size_t)(kt * 64 + key0 + i) * LD + d];
            v8bf s0, s1;
#pragma unroll
            for (int i = 0; i < 8; ++i) { s0[i] = tmp[i]; s1[i] = tmp[i + 8]; }
            *(v8bf*)(ldsV + (size_t)d * ATTN_PITCH + key0)     = s0;
            *(v8bf*)(ldsV + (size_t)d * ATTN_PITCH + key0 + 8) = s1;
        }
        // K tile [64 keys x 64 d] via TDM (natural major: it IS the B operand)
        if (wave == 0) {
            tdm_load_2d(ldsK_off, kptr + (size_t)(kt * 64) * LD, 64, 64,
                        (unsigned long long)LD, /*32 DW rows*/ 4);
            __builtin_amdgcn_s_wait_tensorcnt((short)0);
        }
        __syncthreads();

        // S = Q_wave(16x64) @ K^T
        v8f sacc[4];
#pragma unroll
        for (int nt = 0; nt < 4; ++nt) sacc[nt] = zero8();
#pragma unroll
        for (int kc = 0; kc < 2; ++kc) {
            v16bf aq = load_afrag(ldsQ, wave * 16, ATTN_PITCH, kc * 32);
#pragma unroll
            for (int nt = 0; nt < 4; ++nt) {
                v16bf bk = load_bfrag(ldsK, nt * 16, ATTN_PITCH, kc * 32);
                sacc[nt] = wmma_bf16(aq, bk, sacc[nt]);
            }
        }

        // online softmax (row = i + 8*half within wave tile)
        bf16* myP = ldsP + (size_t)wave * 16 * ATTN_PITCH;
#pragma unroll
        for (int i = 0; i < 8; ++i) {
            const int qg = qbase + wave * 16 + i + 8 * half;
            float sv[4], tmax = -1e30f;
#pragma unroll
            for (int nt = 0; nt < 4; ++nt) {
                const int kg = kt * 64 + nt * 16 + ln;
                float s = sacc[nt][i] * scale;
                if (kg > qg) s = -1e30f;
                sv[nt] = s;
                tmax = fmaxf(tmax, s);
            }
#pragma unroll
            for (int off = 1; off < 16; off <<= 1)
                tmax = fmaxf(tmax, __shfl_xor(tmax, off, 32));
            const float mnew = fmaxf(m_i[i], tmax);
            const float fac  = __expf(m_i[i] - mnew);
            float lsum = 0.0f;
#pragma unroll
            for (int nt = 0; nt < 4; ++nt) {
                const float p = __expf(sv[nt] - mnew);
                lsum += p;
                myP[(size_t)(i + 8 * half) * ATTN_PITCH + nt * 16 + ln] = (bf16)p;
            }
#pragma unroll
            for (int off = 1; off < 16; off <<= 1)
                lsum += __shfl_xor(lsum, off, 32);
            l_i[i] = l_i[i] * fac + lsum;
            m_i[i] = mnew;
#pragma unroll
            for (int dt = 0; dt < 4; ++dt) accO[dt][i] *= fac;
        }

        // O += P(16x64) @ V(64x64)  (wave-private LDS; DS is in-order per wave)
#pragma unroll
        for (int kc = 0; kc < 2; ++kc) {
            v16bf ap = load_afrag(myP, 0, ATTN_PITCH, kc * 32);
#pragma unroll
            for (int dt = 0; dt < 4; ++dt) {
                v16bf bv = load_bfrag(ldsV, dt * 16, ATTN_PITCH, kc * 32);
                accO[dt] = wmma_bf16(ap, bv, accO[dt]);
            }
        }
    }

    bf16* obase = out + (size_t)(b * T + qbase) * 1024 + h * 64;
#pragma unroll
    for (int dt = 0; dt < 4; ++dt) {
#pragma unroll
        for (int i = 0; i < 8; ++i) {
            const int r = wave * 16 + i + 8 * half;
            obase[(size_t)r * 1024 + dt * 16 + ln] = (bf16)(accO[dt][i] / l_i[i]);
        }
    }
}

// ---------------------------------------------------------------------------
// Host launcher
// ---------------------------------------------------------------------------
extern "C" void kernel_launch(void* const* d_in, const int* in_sizes, int n_in,
                              void* d_out, int out_size, void* d_ws, size_t ws_size,
                              hipStream_t stream) {
    (void)in_sizes; (void)n_in; (void)out_size; (void)ws_size;
    const int M = 8192;   // B*T
    const int C = 1024;

    const float* x   = (const float*)d_in[0];
    const float* wq  = (const float*)d_in[1];
    const float* bq  = (const float*)d_in[2];
    const float* wk  = (const float*)d_in[3];
    const float* bk  = (const float*)d_in[4];
    const float* wv  = (const float*)d_in[5];
    const float* bv  = (const float*)d_in[6];
    const float* wp  = (const float*)d_in[7];
    const float* bp  = (const float*)d_in[8];
    const float* w1  = (const float*)d_in[9];
    const float* b1  = (const float*)d_in[10];
    const float* w2  = (const float*)d_in[11];
    const float* b2  = (const float*)d_in[12];
    const float* g1  = (const float*)d_in[13];
    const float* be1 = (const float*)d_in[14];
    const float* g2  = (const float*)d_in[15];
    const float* be2 = (const float*)d_in[16];

    char* p = (char*)d_ws;
    auto alloc = [&](size_t bytes) -> char* {
        char* r = p;
        p += (bytes + 255) & ~(size_t)255;
        return r;
    };
    bf16*  h_bf    = (bf16*) alloc((size_t)M * C * 2);
    bf16*  wqkvT   = (bf16*) alloc((size_t)3072 * C * 2);   // [n][c]
    float* bqkv    = (float*)alloc(3072 * 4);
    bf16*  qkv_bf  = (bf16*) alloc((size_t)M * 3072 * 2);
    bf16*  attn_bf = (bf16*) alloc((size_t)M * C * 2);
    bf16*  wpT     = (bf16*) alloc((size_t)C * C * 2);      // [n][c]
    float* x1      = (float*)alloc((size_t)M * C * 4);
    bf16*  h2_bf   = (bf16*) alloc((size_t)M * C * 2);
    bf16*  w1T     = (bf16*) alloc((size_t)4096 * C * 2);   // [n=4096][k=1024]
    bf16*  ffn1_bf = (bf16*) alloc((size_t)M * 4096 * 2);
    bf16*  w2T     = (bf16*) alloc((size_t)C * 4096 * 2);   // [n=1024][k=4096]

    // 1) weight conversions (bf16, N-major for TDM B-tiles)
    cvt_qkv_kernel<<<(3072 * C) / 256, 256, 0, stream>>>(wq, wk, wv, bq, bk, bv,
                                                         wqkvT, bqkv);
    cvt_bf16T_kernel<<<(C * C) / 256, 256, 0, stream>>>(wp, wpT, C, C);
    cvt_bf16T_kernel<<<(C * 4096) / 256, 256, 0, stream>>>(w1, w1T, 4096, C);
    cvt_bf16T_kernel<<<(4096 * C) / 256, 256, 0, stream>>>(w2, w2T, C, 4096);

    // 2) LN1 -> bf16
    ln_bf16_kernel<<<M, 256, 0, stream>>>(x, g1, be1, h_bf);

    // 3) fused QKV projection: [8192,1024] x [1024,3072]
    gemm_bf16_kernel<true, false, false>
        <<<dim3(M / 128, 3072 / 256), 256, 0, stream>>>(
            h_bf, wqkvT, bqkv, nullptr, qkv_bf, M, 3072, C);

    // 4) causal flash attention -> bf16 [8192,1024]
    attn_kernel<<<dim3(2048 / 128, 64), 256, 0, stream>>>(qkv_bf, attn_bf);

    // 5) output projection + residual -> fp32 x1
    gemm_bf16_kernel<false, false, true>
        <<<dim3(M / 128, C / 256), 256, 0, stream>>>(
            attn_bf, wpT, bp, x, x1, M, C, C);

    // 6) LN2 -> bf16
    ln_bf16_kernel<<<M, 256, 0, stream>>>(x1, g2, be2, h2_bf);

    // 7) FFN1 + ReLU -> bf16 [8192,4096]
    gemm_bf16_kernel<true, true, false>
        <<<dim3(M / 128, 4096 / 256), 256, 0, stream>>>(
            h2_bf, w1T, b1, nullptr, ffn1_bf, M, 4096, C);

    // 8) FFN2 + residual -> fp32 d_out
    gemm_bf16_kernel<false, false, true>
        <<<dim3(M / 128, C / 256), 256, 0, stream>>>(
            ffn1_bf, w2T, b2, x1, (float*)d_out, M, C, 4096);
}
